// STattention_53077205844450
// MI455X (gfx1250) — matrix operand order repaired and checked by reading, not compiled, not run on hardware
//
#include <hip/hip_runtime.h>
#include <hip/hip_bf16.h>

// ---------- Problem constants ----------
#define BB   16
#define CC   256
#define NN   2304      // 48*48
#define HDS  8
#define DD   32
#define N16  144       // NN/16

typedef __attribute__((ext_vector_type(16))) __bf16 v16bf;
typedef __attribute__((ext_vector_type(8)))  __bf16 v8bf;
typedef __attribute__((ext_vector_type(8)))  float  v8f;

// ---------- bf16 helpers (RNE) ----------
__device__ __forceinline__ unsigned short bfbits(float f) {
  unsigned u = __builtin_bit_cast(unsigned, f);
  unsigned r = (u + 0x7FFFu + ((u >> 16) & 1u)) >> 16;
  return (unsigned short)r;
}
__device__ __forceinline__ __bf16 tobf(float f) {
  unsigned short h = bfbits(f);
  return __builtin_bit_cast(__bf16, h);
}
__device__ __forceinline__ unsigned pack2bf(float a, float b) {
  return (unsigned)bfbits(a) | ((unsigned)bfbits(b) << 16);
}

// A-operand: 16x32 bf16 tile, lane L holds row M=L%16, K runs
// {(L/16)*16 .. +16} contiguous at every call site (same k order as the B loads)
__device__ __forceinline__ v16bf load_a16(const __bf16* p0, const __bf16* p1) {
  v8bf lo = *(const v8bf*)p0;
  v8bf hi = *(const v8bf*)p1;
  v16bf r;
#pragma unroll
  for (int i = 0; i < 8; ++i) { r[i] = lo[i]; r[i + 8] = hi[i]; }
  return r;
}
#define WMMA_BF16(a, b, c) \
  __builtin_amdgcn_wmma_f32_16x16x32_bf16(false, (a), false, (b), (short)0, (c), false, false)

// ---------- K0: fp32 -> bf16 weight cast ----------
__global__ __launch_bounds__(256) void k_cast_bf16(const float* __restrict__ s,
                                                   __bf16* __restrict__ d, int n) {
  int i = blockIdx.x * 256 + threadIdx.x;
  if (i < n) d[i] = tobf(s[i]);
}

// ---------- K1: proj1 (1x1 conv GEMM) + bias + LayerNorm + *2 -> t2 bf16 [B,N,C] ----------
__global__ __launch_bounds__(256) void k_proj1_ln(
    const float* __restrict__ x, const __bf16* __restrict__ w1b,
    const float* __restrict__ pb, const float* __restrict__ lng,
    const float* __restrict__ lnb, __bf16* __restrict__ t2) {
  __shared__ alignas(32) __bf16 As[16][CC];   // A = x^T tile [n][c], 8KB
  __shared__ float Os[16][CC];                // GEMM out, 16KB
  __shared__ float Red[16][32];
  const int b  = blockIdx.x / N16;
  const int n0 = (blockIdx.x % N16) * 16;
  const int t  = threadIdx.x;
  const int wv = t >> 5, lane = t & 31, lrow = lane & 15, lhi = lane >> 4;

  { // stage A: transpose x[b][c][n] -> As[n][c], converting to bf16
    const float* xb = x + (size_t)b * CC * NN;
    int ni = t & 15, cb = t >> 4;
#pragma unroll
    for (int j = 0; j < 16; ++j) {
      int c = cb + 16 * j;
      As[ni][c] = tobf(xb[(size_t)c * NN + n0 + ni]);
    }
  }
  __syncthreads();

  v8f acc0 = {}, acc1 = {};
#pragma unroll
  for (int kk = 0; kk < 8; ++kk) {
    int c0 = kk * 32;
    v16bf a = load_a16(&As[lrow][c0 + lhi * 16], &As[lrow][c0 + lhi * 16 + 8]);
    const __bf16* br0 = w1b + (size_t)(wv * 32 + lrow) * CC + c0 + lhi * 16;
    v16bf b0 = *(const v16bf*)br0;
    v16bf b1 = *(const v16bf*)(br0 + 16 * CC);
    acc0 = WMMA_BF16(a, b0, acc0);
    acc1 = WMMA_BF16(a, b1, acc1);
  }
  { // D -> LDS with bias
    int o0 = wv * 32 + lrow;
    float bb0 = pb[o0], bb1 = pb[o0 + 16];
#pragma unroll
    for (int r = 0; r < 8; ++r) {
      int m = r + 8 * lhi;
      Os[m][o0]      = acc0[r] + bb0;
      Os[m][o0 + 16] = acc1[r] + bb1;
    }
  }
  __syncthreads();

  // LayerNorm over C per row, then *2, write bf16
  int row = t >> 4, seg = t & 15;
  float s = 0.f, s2 = 0.f;
#pragma unroll
  for (int j = 0; j < 16; ++j) {
    float v = Os[row][seg * 16 + j];
    s += v; s2 += v * v;
  }
  Red[row][seg] = s; Red[row][16 + seg] = s2;
  __syncthreads();
  float su = 0.f, sq = 0.f;
#pragma unroll
  for (int j = 0; j < 16; ++j) { su += Red[row][j]; sq += Red[row][16 + j]; }
  float mu  = su * (1.f / 256.f);
  float var = sq * (1.f / 256.f) - mu * mu;
  float inv = rsqrtf(var + 1e-5f);
  __bf16* drow = t2 + ((size_t)b * NN + n0 + row) * CC;
#pragma unroll
  for (int j = 0; j < 16; j += 2) {
    int c = seg * 16 + j;
    float v0 = ((Os[row][c]     - mu) * inv * lng[c]     + lnb[c])     * 2.f;
    float v1 = ((Os[row][c + 1] - mu) * inv * lng[c + 1] + lnb[c + 1]) * 2.f;
    *(unsigned*)(drow + c) = pack2bf(v0, v1);
  }
}

// ---------- K2: qkv GEMM -> qb bf16, kf f32, vf f32 (all [B,h,N,d]) ----------
__global__ __launch_bounds__(256) void k_qkv(
    const __bf16* __restrict__ t2, const __bf16* __restrict__ wqb,
    __bf16* __restrict__ qb, float* __restrict__ kf, float* __restrict__ vf) {
  __shared__ alignas(32) __bf16 As[16][CC];
  const int b  = blockIdx.x / N16;
  const int n0 = (blockIdx.x % N16) * 16;
  const int t  = threadIdx.x;
  const int wv = t >> 5, lane = t & 31, lrow = lane & 15, lhi = lane >> 4;

  { // stage A tile (row-major bf16, straight copy)
    int r = t >> 4, c16 = (t & 15) * 16;
    const uint4* src = (const uint4*)(t2 + ((size_t)b * NN + n0 + r) * CC + c16);
    uint4* dst = (uint4*)&As[r][c16];
    dst[0] = src[0]; dst[1] = src[1];
  }
  __syncthreads();

  v8f acc[6] = {};
#pragma unroll
  for (int kk = 0; kk < 8; ++kk) {
    int c0 = kk * 32;
    v16bf a = load_a16(&As[lrow][c0 + lhi * 16], &As[lrow][c0 + lhi * 16 + 8]);
#pragma unroll
    for (int j = 0; j < 6; ++j) {
      int ot = wv * 6 + j;
      const __bf16* br = wqb + (size_t)(ot * 16 + lrow) * CC + c0 + lhi * 16;
      acc[j] = WMMA_BF16(a, *(const v16bf*)br, acc[j]);
    }
  }
  // scatter to q/k/v  (o = three*256 + head*32 + dd)
#pragma unroll
  for (int j = 0; j < 6; ++j) {
    int ot = wv * 6 + j;
    int ob = ot * 16;
    int three = ob >> 8;
    int head  = (ob & 255) >> 5;
    int dd    = (ob & 31) + lrow;
#pragma unroll
    for (int r = 0; r < 8; ++r) {
      int n = n0 + r + 8 * lhi;
      size_t idx = (((size_t)(b * HDS + head)) * NN + n) * DD + dd;
      float v = acc[j][r];
      if (three == 0)      qb[idx] = tobf(v);
      else if (three == 1) kf[idx] = v;
      else                 vf[idx] = v;
    }
  }
}

// ---------- K3a: per-(b,h,dd) softmax stats over N ----------
__global__ __launch_bounds__(256) void k_softstats(
    const float* __restrict__ kf, float* __restrict__ cmax, float* __restrict__ csum) {
  __shared__ float R[8][32];
  int bh = blockIdx.x;
  int t = threadIdx.x, dd = t & 31, ch = t >> 5;
  const float* base = kf + (size_t)bh * NN * DD;
  float m = -3.402823e38f;
  for (int n = ch; n < NN; n += 8) m = fmaxf(m, base[(size_t)n * DD + dd]);
  R[ch][dd] = m;
  __syncthreads();
  float mm = R[0][dd];
#pragma unroll
  for (int j = 1; j < 8; ++j) mm = fmaxf(mm, R[j][dd]);
  __syncthreads();
  float s = 0.f;
  for (int n = ch; n < NN; n += 8) s += __expf(base[(size_t)n * DD + dd] - mm);
  R[ch][dd] = s;
  __syncthreads();
  if (ch == 0) {
    float ss = 0.f;
#pragma unroll
    for (int j = 0; j < 8; ++j) ss += R[j][dd];
    cmax[bh * DD + dd] = mm;
    csum[bh * DD + dd] = ss;
  }
}

// ---------- K3b: kv = softmax(k)^T v  (32x32 per (b,h)), WMMA over K=N ----------
__global__ __launch_bounds__(128) void k_kv(
    const float* __restrict__ kf, const float* __restrict__ vf,
    const float* __restrict__ cmax, const float* __restrict__ csum,
    float* __restrict__ kvf, __bf16* __restrict__ kvT) {
  __shared__ alignas(32) __bf16 As[4][32][32];  // A = ksm^T [kk][n]
  __shared__ alignas(32) __bf16 Bs[4][32][32];  // B^T = v^T [vv][n]
  __shared__ float Acc[4][32][32];
  __shared__ float Ms[32], Si[32];
  int bh = blockIdx.x;
  int t = threadIdx.x, wv = t >> 5, lane = t & 31, lrow = lane & 15, lhi = lane >> 4;
  if (t < 32) { Ms[t] = cmax[bh * DD + t]; Si[t] = 1.f / csum[bh * DD + t]; }
  __syncthreads();

  v8f a00 = {}, a01 = {}, a10 = {}, a11 = {};
  const float* kb = kf + (size_t)bh * NN * DD;
  const float* vb = vf + (size_t)bh * NN * DD;
  for (int s = 0; s < 18; ++s) {                  // 4 waves * 18 = 72 chunks of 32
    int nb = (wv + 4 * s) * 32;
    const float* krow = kb + (size_t)(nb + lane) * DD;
    const float* vrow = vb + (size_t)(nb + lane) * DD;
#pragma unroll
    for (int dd = 0; dd < 32; ++dd) {             // transpose-stage into LDS
      As[wv][dd][lane] = tobf(__expf(krow[dd] - Ms[dd]) * Si[dd]);
      Bs[wv][dd][lane] = tobf(vrow[dd]);
    }
    v16bf aA0 = load_a16(&As[wv][lrow][lhi * 16],      &As[wv][lrow][lhi * 16 + 8]);
    v16bf aA1 = load_a16(&As[wv][16 + lrow][lhi * 16], &As[wv][16 + lrow][lhi * 16 + 8]);
    v16bf bB0 = *(const v16bf*)&Bs[wv][lrow][lhi * 16];
    v16bf bB1 = *(const v16bf*)&Bs[wv][16 + lrow][lhi * 16];
    a00 = WMMA_BF16(aA0, bB0, a00);
    a01 = WMMA_BF16(aA0, bB1, a01);
    a10 = WMMA_BF16(aA1, bB0, a10);
    a11 = WMMA_BF16(aA1, bB1, a11);
  }
#pragma unroll
  for (int r = 0; r < 8; ++r) {
    int m = r + 8 * lhi;
    Acc[wv][m][lrow]           = a00[r];
    Acc[wv][m][16 + lrow]      = a01[r];
    Acc[wv][16 + m][lrow]      = a10[r];
    Acc[wv][16 + m][16 + lrow] = a11[r];
  }
  __syncthreads();
  for (int e = t; e < 1024; e += 128) {
    int kk = e >> 5, vv = e & 31;
    float sv = Acc[0][kk][vv] + Acc[1][kk][vv] + Acc[2][kk][vv] + Acc[3][kk][vv];
    kvf[(size_t)bh * 1024 + e] = sv;
    kvT[(size_t)bh * 1024 + vv * 32 + kk] = tobf(sv);   // store transposed for B-operand use
  }
}

// ---------- K4: SE gate MLP + n_sub (transposed bf16) ----------
__global__ __launch_bounds__(256) void k_se(
    const float* __restrict__ kvf,
    const float* __restrict__ fc1w, const float* __restrict__ fc1b,
    const float* __restrict__ bng, const float* __restrict__ bnb,
    const float* __restrict__ bnm, const float* __restrict__ bnv,
    const float* __restrict__ fc2w, const float* __restrict__ fc2b,
    __bf16* __restrict__ nsubT) {
  __shared__ float avgS[256], y1S[256], gateS[256];
  int b = blockIdx.x, t = threadIdx.x;
  int h = t >> 5, vv = t & 31;
  const float* kvb = kvf + (size_t)b * HDS * 1024;
  float s = 0.f;
#pragma unroll
  for (int kk = 0; kk < 32; ++kk) s += fabsf(kvb[h * 1024 + kk * 32 + vv]);
  avgS[t] = s * (1.f / 32.f);
  __syncthreads();
  { // fc1 + BN(eval) + relu
    const float* wr = fc1w + (size_t)t * 256;
    float acc = fc1b[t];
    for (int c = 0; c < 256; ++c) acc += avgS[c] * wr[c];
    acc = (acc - bnm[t]) * rsqrtf(bnv[t] + 1e-5f) * bng[t] + bnb[t];
    y1S[t] = fmaxf(acc, 0.f);
  }
  __syncthreads();
  { // fc2 + sigmoid, gate
    const float* wr = fc2w + (size_t)t * 256;
    float acc = fc2b[t];
    for (int c = 0; c < 256; ++c) acc += y1S[c] * wr[c];
    gateS[t] = avgS[t] * (1.f / (1.f + __expf(-acc)));
  }
  __syncthreads();
  // nsubT[b,h][vv][kk] = relu(|kv[kk][vv]| - gate[h,vv])
  float g = gateS[t];
  __bf16* dr = nsubT + ((size_t)(b * HDS + h) * 32 + vv) * 32;
#pragma unroll
  for (int kk = 0; kk < 32; ++kk) {
    float nv = fabsf(kvb[h * 1024 + kk * 32 + vv]) - g;
    dr[kk] = tobf(fmaxf(nv, 0.f));
  }
}

// ---------- K5: fa = q @ kv ; out = (sign(fa) @ n_sub + v) * scale -> attn bf16 [B,N,C] ----------
__global__ __launch_bounds__(256) void k_attnout(
    const __bf16* __restrict__ qb, const __bf16* __restrict__ kvT,
    const __bf16* __restrict__ nsubT, const float* __restrict__ vf,
    __bf16* __restrict__ attn) {
  __shared__ alignas(32) __bf16 Ssgn[8][16][32];
  int bh = blockIdx.x / 18, tb = blockIdx.x % 18;
  int b = bh >> 3, h = bh & 7;
  int t = threadIdx.x, wv = t >> 5, lane = t & 31, lrow = lane & 15, lhi = lane >> 4;
  int n0 = (tb * 8 + wv) * 16;

  const __bf16* qrow = qb + ((size_t)bh * NN + n0 + lrow) * DD;
  v16bf aq = load_a16(qrow + lhi * 16, qrow + lhi * 16 + 8);
  const __bf16* kvb = kvT + (size_t)bh * 1024;
  v16bf bk0 = *(const v16bf*)(kvb + (size_t)lrow * 32 + lhi * 16);
  v16bf bk1 = *(const v16bf*)(kvb + (size_t)(16 + lrow) * 32 + lhi * 16);
  v8f z = {};
  v8f fa0 = WMMA_BF16(aq, bk0, z);
  v8f fa1 = WMMA_BF16(aq, bk1, z);
#pragma unroll
  for (int r = 0; r < 8; ++r) {
    int m = r + 8 * lhi;
    float f0 = fa0[r], f1 = fa1[r];
    Ssgn[wv][m][lrow]      = tobf(f0 > 0.f ? 1.f : (f0 < 0.f ? -1.f : 0.f));
    Ssgn[wv][m][16 + lrow] = tobf(f1 > 0.f ? 1.f : (f1 < 0.f ? -1.f : 0.f));
  }
  __syncthreads();
  v16bf as = load_a16(&Ssgn[wv][lrow][lhi * 16], &Ssgn[wv][lrow][lhi * 16 + 8]);
  const __bf16* nsb = nsubT + (size_t)bh * 1024;
  v16bf bn0 = *(const v16bf*)(nsb + (size_t)lrow * 32 + lhi * 16);
  v16bf bn1 = *(const v16bf*)(nsb + (size_t)(16 + lrow) * 32 + lhi * 16);
  v8f o0 = WMMA_BF16(as, bn0, z);
  v8f o1 = WMMA_BF16(as, bn1, z);
  const float scl = 0.17677669529663687f;  // d^-0.5
#pragma unroll
  for (int r = 0; r < 8; ++r) {
    int n = n0 + r + 8 * lhi;
    const float* vr = vf + ((size_t)bh * NN + n) * DD;
    __bf16* ar = attn + ((size_t)b * NN + n) * CC + h * DD;
    ar[lrow]      = tobf((o0[r] + vr[lrow]) * scl);
    ar[16 + lrow] = tobf((o1[r] + vr[16 + lrow]) * scl);
  }
}

// ---------- K6: proj3 GEMM + bias, transposed f32 output [B,C,N] ----------
__global__ __launch_bounds__(256) void k_proj3(
    const __bf16* __restrict__ attn, const __bf16* __restrict__ w3b,
    const float* __restrict__ p3b, float* __restrict__ out) {
  __shared__ alignas(32) __bf16 As[16][CC];
  __shared__ float Os[16][CC];
  int b = blockIdx.x / N16, n0 = (blockIdx.x % N16) * 16;
  int t = threadIdx.x, wv = t >> 5, lane = t & 31, lrow = lane & 15, lhi = lane >> 4;
  {
    int r = t >> 4, c16 = (t & 15) * 16;
    const uint4* src = (const uint4*)(attn + ((size_t)b * NN + n0 + r) * CC + c16);
    uint4* dst = (uint4*)&As[r][c16];
    dst[0] = src[0]; dst[1] = src[1];
  }
  __syncthreads();
  v8f acc0 = {}, acc1 = {};
#pragma unroll
  for (int kk = 0; kk < 8; ++kk) {
    int c0 = kk * 32;
    v16bf a = load_a16(&As[lrow][c0 + lhi * 16], &As[lrow][c0 + lhi * 16 + 8]);
    const __bf16* br0 = w3b + (size_t)(wv * 32 + lrow) * CC + c0 + lhi * 16;
    v16bf b0 = *(const v16bf*)br0;
    v16bf b1 = *(const v16bf*)(br0 + 16 * CC);
    acc0 = WMMA_BF16(a, b0, acc0);
    acc1 = WMMA_BF16(a, b1, acc1);
  }
  {
    int o0 = wv * 32 + lrow;
    float bb0 = p3b[o0], bb1 = p3b[o0 + 16];
#pragma unroll
    for (int r = 0; r < 8; ++r) {
      int m = r + 8 * lhi;
      Os[m][o0]      = acc0[r] + bb0;
      Os[m][o0 + 16] = acc1[r] + bb1;
    }
  }
  __syncthreads();
  { // transposed write: out[b][o][n0..n0+16)
    int o = t;
    float* dr = out + ((size_t)b * CC + o) * NN + n0;
#pragma unroll
    for (int q = 0; q < 4; ++q) {
      float4 vw = make_float4(Os[q * 4 + 0][o], Os[q * 4 + 1][o],
                              Os[q * 4 + 2][o], Os[q * 4 + 3][o]);
      *(float4*)(dr + q * 4) = vw;
    }
  }
}

// ---------- Host launcher ----------
extern "C" void kernel_launch(void* const* d_in, const int* in_sizes, int n_in,
                              void* d_out, int out_size, void* d_ws, size_t ws_size,
                              hipStream_t stream) {
  (void)in_sizes; (void)n_in; (void)out_size; (void)ws_size;
  const float* x    = (const float*)d_in[0];
  const float* w1   = (const float*)d_in[1];
  const float* p1b  = (const float*)d_in[2];
  const float* lng  = (const float*)d_in[3];
  const float* lnb  = (const float*)d_in[4];
  const float* wqkv = (const float*)d_in[5];
  const float* fc1w = (const float*)d_in[6];
  const float* fc1b = (const float*)d_in[7];
  const float* bng  = (const float*)d_in[8];
  const float* bnb  = (const float*)d_in[9];
  const float* bnm  = (const float*)d_in[10];
  const float* bnv  = (const float*)d_in[11];
  const float* fc2w = (const float*)d_in[12];
  const float* fc2b = (const float*)d_in[13];
  const float* w3   = (const float*)d_in[14];
  const float* p3b  = (const float*)d_in[15];
  float* out = (float*)d_out;

  char* ws = (char*)d_ws;
  size_t off = 0;
  auto alloc = [&](size_t bytes) {
    size_t r = off;
    off = (off + bytes + 255) & ~(size_t)255;
    return r;
  };
  __bf16* w1b   = (__bf16*)(ws + alloc((size_t)CC * CC * 2));
  __bf16* wqb   = (__bf16*)(ws + alloc((size_t)3 * CC * CC * 2));
  __bf16* w3b   = (__bf16*)(ws + alloc((size_t)CC * CC * 2));
  __bf16* t2    = (__bf16*)(ws + alloc((size_t)BB * NN * CC * 2));
  float*  kf    = (float*) (ws + alloc((size_t)BB * HDS * NN * DD * 4));
  float*  vfp   = (float*) (ws + alloc((size_t)BB * HDS * NN * DD * 4));
  __bf16* qbp   = (__bf16*)(ws + alloc((size_t)BB * HDS * NN * DD * 2));
  float*  cmax  = (float*) (ws + alloc((size_t)BB * HDS * DD * 4));
  float*  csum  = (float*) (ws + alloc((size_t)BB * HDS * DD * 4));
  float*  kvf   = (float*) (ws + alloc((size_t)BB * HDS * DD * DD * 4));
  __bf16* kvT   = (__bf16*)(ws + alloc((size_t)BB * HDS * DD * DD * 2));
  __bf16* nsubT = (__bf16*)(ws + alloc((size_t)BB * HDS * DD * DD * 2));
  __bf16* attn  = (__bf16*)(ws + alloc((size_t)BB * NN * CC * 2));

  k_cast_bf16<<<(CC * CC + 255) / 256, 256, 0, stream>>>(w1, w1b, CC * CC);
  k_cast_bf16<<<(3 * CC * CC + 255) / 256, 256, 0, stream>>>(wqkv, wqb, 3 * CC * CC);
  k_cast_bf16<<<(CC * CC + 255) / 256, 256, 0, stream>>>(w3, w3b, CC * CC);

  k_proj1_ln<<<BB * N16, 256, 0, stream>>>(x, w1b, p1b, lng, lnb, t2);
  k_qkv<<<BB * N16, 256, 0, stream>>>(t2, wqb, qbp, kf, vfp);
  k_softstats<<<BB * HDS, 256, 0, stream>>>(kf, cmax, csum);
  k_kv<<<BB * HDS, 128, 0, stream>>>(kf, vfp, cmax, csum, kvf, kvT);
  k_se<<<BB, 256, 0, stream>>>(kvf, fc1w, fc1b, bng, bnb, bnm, bnv, fc2w, fc2b, nsubT);
  k_attnout<<<BB * HDS * 18, 256, 0, stream>>>(qbp, kvT, nsubT, vfp, attn);
  k_proj3<<<BB * N16, 256, 0, stream>>>(attn, w3b, p3b, out);
}
